// GRNN_65944927863125
// MI455X (gfx1250) — matrix operand order repaired and checked by reading, not compiled, not run on hardware
//
#include <hip/hip_runtime.h>
#include <hip/hip_bf16.h>

typedef __attribute__((ext_vector_type(16))) __bf16       v16bf;
typedef __attribute__((ext_vector_type(4)))  __bf16       v4bf;
typedef __attribute__((ext_vector_type(8)))  float        v8f;
typedef __attribute__((ext_vector_type(4)))  unsigned int v4u;
typedef __attribute__((ext_vector_type(8)))  int          v8i;
typedef __attribute__((ext_vector_type(4)))  int          v4i;

#define GRNN_SIGMA 0.2f
#define MSLICES 4        // grid.y: independent M-range slices (deterministic partials)

// ---------------------------------------------------------------------------
// TDM: 1-D linear DMA of `ndw` dwords from global memory into LDS (ISA ch.8).
// Group0: count=1 | lds_addr | global_addr[56:0] | type=2 ("image").
// Group1: data_size=4B (code 2); tensor_dim0 = tile_dim0 = ndw; tile_dim1/2 = 0
//         (1-D tile); stride = ndw (unused for 1-D); no pad/iterate/multicast.
// ---------------------------------------------------------------------------
__device__ __forceinline__ void grnn_tdm_load_1d(unsigned lds_off,
                                                 const void* gptr,
                                                 unsigned ndw) {
  const unsigned long long ga = (unsigned long long)gptr;
  v4u g0;
  g0[0] = 1u;                                                // count=1
  g0[1] = lds_off;                                           // lds_addr (bytes)
  g0[2] = (unsigned)(ga & 0xFFFFFFFFu);                      // addr[31:0]
  g0[3] = (unsigned)((ga >> 32) & 0x01FFFFFFu) | (2u << 30); // addr[56:32]|type=2
  v8i g1;
  g1[0] = (int)(2u << 16);                  // data_size = 4 bytes
  g1[1] = (int)((ndw & 0xFFFFu) << 16);     // tensor_dim0[15:0]
  g1[2] = (int)((ndw >> 16) & 0xFFFFu);     // tensor_dim0[31:16]; tensor_dim1=0
  g1[3] = (int)((ndw & 0xFFFFu) << 16);     // tile_dim0 = ndw (16-bit)
  g1[4] = 0;                                // tile_dim1 = tile_dim2 = 0 (1-D)
  g1[5] = (int)ndw;                         // tensor_dim0_stride
  g1[6] = 0;
  g1[7] = 0;
  const v4i z = {0, 0, 0, 0};
#if __clang_major__ >= 23
  const v8i z8 = {0, 0, 0, 0, 0, 0, 0, 0};
  __builtin_amdgcn_tensor_load_to_lds(g0, g1, z, z, z8, 0);
#else
  __builtin_amdgcn_tensor_load_to_lds(g0, g1, z, z, 0);
#endif
}

// ---------------------------------------------------------------------------
// Kernel 1: per-row squared norms.  One wave (32 lanes) per row of 256 floats.
// ---------------------------------------------------------------------------
__global__ __launch_bounds__(128) void grnn_sqnorm_kernel(
    const float* __restrict__ src, float* __restrict__ dst, int rows) {
  const int wave = threadIdx.x >> 5;
  const int lane = threadIdx.x & 31;
  const int row  = blockIdx.x * 4 + wave;
  if (row >= rows) return;
  const float* p = src + (size_t)row * 256;
  float s = 0.0f;
#pragma unroll
  for (int i = 0; i < 8; ++i) {
    float v = p[lane + 32 * i];
    s += v * v;
  }
#pragma unroll
  for (int m = 16; m >= 1; m >>= 1) s += __shfl_xor(s, m, 32);
  if (lane == 0) dst[row] = s;
}

// ---------------------------------------------------------------------------
// Kernel 2: one-shot f32 -> (bf16 hi | bf16 lo) split of X_train into a single
// interleaved plane XTs[M][512]: per row, elements 0..255 = hi, 256..511 = lo.
// A 16-row tile is therefore one CONTIGUOUS 16KB block -> single 1-D TDM copy.
// ---------------------------------------------------------------------------
__global__ __launch_bounds__(256) void grnn_split_kernel(
    const float* __restrict__ src, __bf16* __restrict__ dst, int n4) {
  const int i = blockIdx.x * 256 + threadIdx.x;
  if (i >= n4) return;
  const float4 v = ((const float4*)src)[i];
  const float f[4] = {v.x, v.y, v.z, v.w};
  const int row = i >> 6;          // 64 float4 per 256-wide row
  const int k4  = (i & 63) * 4;
  v4bf h, l;
#pragma unroll
  for (int t = 0; t < 4; ++t) {
    __bf16 hh = (__bf16)f[t];
    h[t] = hh;
    l[t] = (__bf16)(f[t] - (float)hh);
  }
  *(v4bf*)(dst + (size_t)row * 512 + k4)       = h;
  *(v4bf*)(dst + (size_t)row * 512 + 256 + k4) = l;
}

// ---------------------------------------------------------------------------
// Kernel 3: fused GEMM (bf16x3 -> f32-grade accuracy) + exp + row-reduce.
// Block = 128 thr (4 waves); each wave owns a 16-row n-tile (A resident in
// VGPRs).  B tiles are DMA'd ONCE per block into double-buffered LDS by the
// TDM (wave 0), all 4 waves consume via ds_load_b128 -> 4x less L2 traffic
// than per-wave global loads.  grid.y slices the M range 4x for occupancy;
// per-slice partial (num, den) go to ws, summed deterministically in kernel 4.
//
// WMMA operand layouts used here (wave32; A/B k order is contiguous, not the ISA table order):
//   A 16x32 bf16 : lane L holds row M=L&15; halves 0..15 = K(kb + (L>>4)*16 + t)
//   B 32x16 bf16 : lane L holds col N=L&15; halves 0..15 = K(kb + (L>>4)*16 + t)
//   C 16x16 f32  : lane L, vgpr r -> element (M = r + 8*(L>>4), N = L&15)
// ---------------------------------------------------------------------------
__global__ __launch_bounds__(128) void grnn_main_kernel(
    const float* __restrict__ X, const __bf16* __restrict__ XTs,
    const float* __restrict__ y, const float* __restrict__ xq2,
    const float* __restrict__ xt2, float* __restrict__ npart,
    float* __restrict__ dpart, int N, int Mslice) {
  __shared__ __align__(64) __bf16 smem[2][16 * 512];  // 2 x 16KB tile buffers

  const int lane  = threadIdx.x & 31;
  const int wave  = threadIdx.x >> 5;
  const int nbase = (blockIdx.x * 4 + wave) * 16;
  const int slice = blockIdx.y;

  const int arow  = nbase + (lane & 15);
  const int akoff = (lane >> 4) * 16;
  const int bkoff = (lane >> 4) * 16;
  const int mgrp  = (lane >> 4) * 8;

  // ---- Preload A tile (16 rows x 256 K) as hi/lo bf16 split (once) ----
  v16bf a_hi[8], a_lo[8];
  {
    const float* ap = X + (size_t)arow * 256;
#pragma unroll
    for (int c = 0; c < 8; ++c) {
      const int kb = c * 32 + akoff;
#pragma unroll
      for (int t = 0; t < 8; ++t) {
        float v0 = ap[kb + t];
        float v1 = ap[kb + 8 + t];
        __bf16 h0 = (__bf16)v0;
        __bf16 h1 = (__bf16)v1;
        a_hi[c][t]     = h0;
        a_lo[c][t]     = (__bf16)(v0 - (float)h0);
        a_hi[c][8 + t] = h1;
        a_lo[c][8 + t] = (__bf16)(v1 - (float)h1);
      }
    }
  }

  float xq2v[8];
#pragma unroll
  for (int r = 0; r < 8; ++r) xq2v[r] = xq2[nbase + mgrp + r];

  const float inv2s2 = 1.0f / (2.0f * GRNN_SIGMA * GRNN_SIGMA);
  float accn[8], accd[8];
#pragma unroll
  for (int r = 0; r < 8; ++r) { accn[r] = 0.0f; accd[r] = 0.0f; }

  const int    mbegin   = slice * Mslice;
  const int    ntiles   = Mslice / 16;
  const __bf16* tilebase = XTs + (size_t)mbegin * 512;  // 1KB per row
  const unsigned ldsoff[2] = {(unsigned)(size_t)&smem[0][0],
                              (unsigned)(size_t)&smem[1][0]};

  // Prologue: DMA tile 0 (wave 0 only; TDM ignores EXEC, one op per wave).
  if (threadIdx.x < 32) grnn_tdm_load_1d(ldsoff[0], tilebase, 4096);

  int cur = 0;
  for (int mt = 0; mt < ntiles; ++mt) {
    if (threadIdx.x < 32) __builtin_amdgcn_s_wait_tensorcnt(0);
    __syncthreads();  // current buffer filled, visible to all waves

    if ((mt + 1 < ntiles) && threadIdx.x < 32)  // DMA next tile into other buf
      grnn_tdm_load_1d(ldsoff[cur ^ 1], tilebase + (size_t)(mt + 1) * 16 * 512,
                       4096);

    const int brow   = mbegin + mt * 16 + (lane & 15);
    const float xt2l = xt2[brow];
    const float yl   = y[brow];
    const __bf16* rowp = &smem[cur][(lane & 15) * 512];

    v8f c = {};
#pragma unroll
    for (int cc = 0; cc < 8; ++cc) {
      const int kb = cc * 32 + bkoff;
      const v16bf bhi = *(const v16bf*)(rowp + kb);        // ds_load_b128 x2
      const v16bf blo = *(const v16bf*)(rowp + 256 + kb);
      // bf16x3: hi*hi + hi*lo + lo*hi  (lo*lo ~2^-16 rel., dropped)
      c = __builtin_amdgcn_wmma_f32_16x16x32_bf16(false, a_hi[cc], false, bhi,
                                                  (short)0, c, false, false);
      c = __builtin_amdgcn_wmma_f32_16x16x32_bf16(false, a_hi[cc], false, blo,
                                                  (short)0, c, false, false);
      c = __builtin_amdgcn_wmma_f32_16x16x32_bf16(false, a_lo[cc], false, bhi,
                                                  (short)0, c, false, false);
    }

#pragma unroll
    for (int r = 0; r < 8; ++r) {
      float d2 = xq2v[r] + xt2l - 2.0f * c[r];  // ||xq - xt||^2 (>= 0)
      float w  = __expf(-d2 * inv2s2);
      accn[r] += w * yl;
      accd[r] += w;
    }

    __syncthreads();  // all waves done reading `cur` before it is re-filled
    cur ^= 1;
  }

  // ---- Reduce over the 16 N-lanes of each half-wave, write slice partials --
#pragma unroll
  for (int r = 0; r < 8; ++r) {
    float n = accn[r], d = accd[r];
#pragma unroll
    for (int m = 8; m >= 1; m >>= 1) {
      n += __shfl_xor(n, m, 32);
      d += __shfl_xor(d, m, 32);
    }
    if ((lane & 15) == 0) {
      npart[(size_t)slice * N + nbase + mgrp + r] = n;
      dpart[(size_t)slice * N + nbase + mgrp + r] = d;
    }
  }
}

// ---------------------------------------------------------------------------
// Kernel 4: deterministic sum of the MSLICES partials, final divide.
// ---------------------------------------------------------------------------
__global__ __launch_bounds__(256) void grnn_finalize_kernel(
    const float* __restrict__ npart, const float* __restrict__ dpart,
    float* __restrict__ out, int N) {
  const int n = blockIdx.x * 256 + threadIdx.x;
  if (n >= N) return;
  float a = 0.0f, b = 0.0f;
#pragma unroll
  for (int s = 0; s < MSLICES; ++s) {
    a += npart[(size_t)s * N + n];
    b += dpart[(size_t)s * N + n];
  }
  out[n] = a / (b + 1e-9f);
}

// ---------------------------------------------------------------------------
extern "C" void kernel_launch(void* const* d_in, const int* in_sizes, int n_in,
                              void* d_out, int out_size, void* d_ws, size_t ws_size,
                              hipStream_t stream) {
  const float* X  = (const float*)d_in[0];   // [N, 256]
  const float* XT = (const float*)d_in[1];   // [M, 256]
  const float* y  = (const float*)d_in[2];   // [M, 1]
  float* out = (float*)d_out;

  const int N = in_sizes[0] / 256;  // 4096
  const int M = in_sizes[1] / 256;  // 8192

  // ws: xq2[N] | xt2[M] | npart[4N] | dpart[4N] | XTs[M*512] bf16  (~8.4 MB)
  float*  xq2   = (float*)d_ws;
  float*  xt2   = xq2 + N;
  float*  npart = xt2 + M;
  float*  dpart = npart + (size_t)MSLICES * N;
  __bf16* XTs   = (__bf16*)(dpart + (size_t)MSLICES * N);

  grnn_sqnorm_kernel<<<(N + 3) / 4, 128, 0, stream>>>(X, xq2, N);
  grnn_sqnorm_kernel<<<(M + 3) / 4, 128, 0, stream>>>(XT, xt2, M);

  const int n4 = (M * 256) / 4;
  grnn_split_kernel<<<(n4 + 255) / 256, 256, 0, stream>>>(XT, XTs, n4);

  dim3 grid(N / 64, MSLICES);  // 64 n-blocks x 4 m-slices = 256 blocks
  grnn_main_kernel<<<grid, 128, 0, stream>>>(X, XTs, y, xq2, xt2, npart, dpart,
                                             N, M / MSLICES);

  grnn_finalize_kernel<<<(N + 255) / 256, 256, 0, stream>>>(npart, dpart, out, N);
}
